// AttModel_2860448219706
// MI455X (gfx1250) — compile-verified, matrix operand order checked by reading, not run on hardware
//
#include <hip/hip_runtime.h>
#include <hip/hip_bf16.h>

typedef __attribute__((ext_vector_type(16))) _Float16 v16h;
typedef __attribute__((ext_vector_type(8)))  _Float16 v8h;
typedef __attribute__((ext_vector_type(8)))  float    v8f;

#define B_    32
#define N_    1024
#define DIN_  256
#define H_    256
#define DOUT_ 256
#define NEGC  9000000000000000.0f

// Load one 16x32 f16 A/B fragment for this lane.
// p points at this lane's K-base element; halves [0..7] at p, [8..15] at p+16.
__device__ __forceinline__ v16h load_frag(const _Float16* p) {
  v8h lo = *(const v8h*)(p);
  v8h hi = *(const v8h*)(p + 16);
  v16h r;
#pragma unroll
  for (int i = 0; i < 8; ++i) { r[i] = lo[i]; r[i + 8] = hi[i]; }
  return r;
}

// ---------------- conversion kernels ----------------

__global__ void cvt_f32_to_f16(const float* __restrict__ src,
                               _Float16* __restrict__ dst, int n) {
  int i = blockIdx.x * blockDim.x + threadIdx.x;
  int stride = gridDim.x * blockDim.x;
  for (; i < n; i += stride) dst[i] = (_Float16)src[i];
}

// w: [DIN][H] row-major (fp32)  ->  wt: [H][DIN] f16 (column-major of w)
__global__ void cvt_transpose_w(const float* __restrict__ w,
                                _Float16* __restrict__ wt) {
  int idx = blockIdx.x * blockDim.x + threadIdx.x;  // 0..65535
  int k = idx >> 8;
  int n = idx & 255;
  wt[n * DIN_ + k] = (_Float16)w[k * H_ + n];
}

// ---------------- fused QKV projection ----------------
// grid: (B*N/16, 3); block 256 (8 waves); wave w -> output col tiles 2w, 2w+1
__global__ void __launch_bounds__(256)
proj_qkv(const _Float16* __restrict__ xh,
         const _Float16* __restrict__ wqh, const float* __restrict__ bq,
         const _Float16* __restrict__ wkh, const float* __restrict__ bk,
         const _Float16* __restrict__ wvh, const float* __restrict__ bv,
         _Float16* __restrict__ qh, _Float16* __restrict__ kh,
         _Float16* __restrict__ vth) {
  const int z = blockIdx.y;
  const _Float16* wt;
  const float* bias;
  if (z == 0)      { wt = wqh; bias = bq; }
  else if (z == 1) { wt = wkh; bias = bk; }
  else             { wt = wvh; bias = bv; }

  const int lane  = threadIdx.x & 31;
  const int wave  = threadIdx.x >> 5;
  const int rt    = blockIdx.x;          // 16-row tile over flat B*N
  const int mrow  = lane & 15;           // A row / B col / C col within tile
  const int kb    = (lane >> 4) * 8;     // K-base within a 32-chunk
  const int rhalf = (lane >> 4) * 8;     // C row offset

  v16h a[8];
  const _Float16* xrow = xh + (size_t)(rt * 16 + mrow) * DIN_;
#pragma unroll
  for (int s = 0; s < 8; ++s) a[s] = load_frag(xrow + s * 32 + kb);

#pragma unroll
  for (int ct2 = 0; ct2 < 2; ++ct2) {
    const int n = (wave * 2 + ct2) * 16 + mrow;   // output column
    v8f c = {};
    const _Float16* wcol = wt + (size_t)n * DIN_;
#pragma unroll
    for (int s = 0; s < 8; ++s) {
      v16h b = load_frag(wcol + s * 32 + kb);
      c = __builtin_amdgcn_wmma_f32_16x16x32_f16(false, a[s], false, b,
                                                 (short)0, c, false, false);
    }
    const float bval = bias[n];
#pragma unroll
    for (int g = 0; g < 8; ++g) {
      const int R = rt * 16 + g + rhalf;          // flat row in [0, B*N)
      float v = c[g] + bval;
      v = v > 0.f ? v : 0.f;
      _Float16 hv = (_Float16)v;
      if (z == 0)      qh[(size_t)R * H_ + n] = hv;
      else if (z == 1) kh[(size_t)R * H_ + n] = hv;
      else {
        const int bb = R >> 10;        // N_ == 1024
        const int nn = R & 1023;
        vth[((size_t)bb * H_ + n) * N_ + nn] = hv;   // transposed store
      }
    }
  }
}

// ---------------- attention: scores -> softmax -> P@V ----------------
// grid: (N/16, B); block 256 (8 waves)
__global__ void __launch_bounds__(256)
attention(const _Float16* __restrict__ qh, const _Float16* __restrict__ kh,
          const _Float16* __restrict__ vth, const float* __restrict__ mask,
          _Float16* __restrict__ ah) {
  __shared__ __attribute__((aligned(16))) float    S[16][N_];   // 64 KB
  __shared__ __attribute__((aligned(16))) _Float16 P[16][N_];   // 32 KB
  __shared__ float red[16][16];                                  // 1 KB

  const int b     = blockIdx.y;
  const int rt    = blockIdx.x;
  const int lane  = threadIdx.x & 31;
  const int wave  = threadIdx.x >> 5;
  const int mrow  = lane & 15;
  const int kb    = (lane >> 4) * 8;
  const int rhalf = (lane >> 4) * 8;

  const size_t qkvBase = (size_t)b * N_ * H_;

  // Q fragments for the 16-row tile (K = H = 256 -> 8 steps)
  v16h qa[8];
  const _Float16* qrow = qh + qkvBase + (size_t)(rt * 16 + mrow) * H_;
#pragma unroll
  for (int s = 0; s < 8; ++s) qa[s] = load_frag(qrow + s * 32 + kb);

  const float* mbase = mask + ((size_t)b * N_ + (size_t)rt * 16) * N_;

  // ---- pass 1: masked scores into LDS (8 column tiles per wave) ----
  for (int t = 0; t < 8; ++t) {
    const int ct = wave * 8 + t;
    const int m  = ct * 16 + mrow;               // key index (B col / C col)
    v8f c = {};
    const _Float16* krow = kh + qkvBase + (size_t)m * H_;
#pragma unroll
    for (int s = 0; s < 8; ++s) {
      v16h bf = load_frag(krow + s * 32 + kb);
      c = __builtin_amdgcn_wmma_f32_16x16x32_f16(false, qa[s], false, bf,
                                                 (short)0, c, false, false);
    }
#pragma unroll
    for (int g = 0; g < 8; ++g) {
      const int r = g + rhalf;                   // row within tile
      const float mv = mbase[(size_t)r * N_ + m];
      S[r][m] = c[g] * mv - NEGC * (1.0f - mv);  // exact reference formula
    }
  }
  __syncthreads();

  // ---- softmax over each of the 16 rows (16 threads per row) ----
  {
    const int row = threadIdx.x >> 4;
    const int t16 = threadIdx.x & 15;
    float mx = -3.4e38f;
#pragma unroll 4
    for (int j = 0; j < 64; ++j) {
      float s = S[row][t16 + 16 * j];
      mx = s > mx ? s : mx;
    }
    red[row][t16] = mx;
    __syncthreads();
#pragma unroll
    for (int i = 0; i < 16; ++i) { float s = red[row][i]; mx = s > mx ? s : mx; }
    __syncthreads();                              // done reading red (max)
    float sm = 0.f;
#pragma unroll 4
    for (int j = 0; j < 64; ++j) {
      float e = __expf(S[row][t16 + 16 * j] - mx);
      S[row][t16 + 16 * j] = e;
      sm += e;
    }
    red[row][t16] = sm;
    __syncthreads();
    float tot = 0.f;
#pragma unroll
    for (int i = 0; i < 16; ++i) tot += red[row][i];
    const float inv = 1.0f / tot;
#pragma unroll 4
    for (int j = 0; j < 64; ++j)
      P[row][t16 + 16 * j] = (_Float16)(S[row][t16 + 16 * j] * inv);
  }
  __syncthreads();

  // ---- pass 2: O = P @ V, K = N = 1024 (32 steps), 2 col tiles per wave ----
#pragma unroll
  for (int ct2 = 0; ct2 < 2; ++ct2) {
    const int h = (wave * 2 + ct2) * 16 + mrow;   // output head-dim column
    v8f c = {};
    const _Float16* vcol = vth + ((size_t)b * H_ + h) * N_;
#pragma unroll 4
    for (int s = 0; s < 32; ++s) {
      v8h lo = *(const v8h*)&P[mrow][s * 32 + kb];        // ds b128
      v8h hi = *(const v8h*)&P[mrow][s * 32 + kb + 16];   // ds b128
      v16h af;
#pragma unroll
      for (int i = 0; i < 8; ++i) { af[i] = lo[i]; af[i + 8] = hi[i]; }
      v16h bf = load_frag(vcol + s * 32 + kb);
      c = __builtin_amdgcn_wmma_f32_16x16x32_f16(false, af, false, bf,
                                                 (short)0, c, false, false);
    }
#pragma unroll
    for (int g = 0; g < 8; ++g) {
      const int R = rt * 16 + g + rhalf;
      ah[qkvBase + (size_t)R * H_ + h] = (_Float16)c[g];
    }
  }
}

// ---------------- output projection (fp32 out, relu) ----------------
__global__ void __launch_bounds__(256)
out_proj(const _Float16* __restrict__ ah, const _Float16* __restrict__ woh,
         const float* __restrict__ bo, float* __restrict__ out) {
  const int lane  = threadIdx.x & 31;
  const int wave  = threadIdx.x >> 5;
  const int rt    = blockIdx.x;
  const int mrow  = lane & 15;
  const int kb    = (lane >> 4) * 8;
  const int rhalf = (lane >> 4) * 8;

  v16h a[8];
  const _Float16* arow = ah + (size_t)(rt * 16 + mrow) * H_;
#pragma unroll
  for (int s = 0; s < 8; ++s) a[s] = load_frag(arow + s * 32 + kb);

#pragma unroll
  for (int ct2 = 0; ct2 < 2; ++ct2) {
    const int n = (wave * 2 + ct2) * 16 + mrow;
    v8f c = {};
    const _Float16* wcol = woh + (size_t)n * H_;
#pragma unroll
    for (int s = 0; s < 8; ++s) {
      v16h bf = load_frag(wcol + s * 32 + kb);
      c = __builtin_amdgcn_wmma_f32_16x16x32_f16(false, a[s], false, bf,
                                                 (short)0, c, false, false);
    }
    const float bval = bo[n];
#pragma unroll
    for (int g = 0; g < 8; ++g) {
      const int R = rt * 16 + g + rhalf;
      float v = c[g] + bval;
      out[(size_t)R * DOUT_ + n] = v > 0.f ? v : 0.f;
    }
  }
}

// ---------------- launch ----------------

extern "C" void kernel_launch(void* const* d_in, const int* in_sizes, int n_in,
                              void* d_out, int out_size, void* d_ws,
                              size_t ws_size, hipStream_t stream) {
  const float* x    = (const float*)d_in[0];
  const float* mask = (const float*)d_in[1];
  const float* Wv   = (const float*)d_in[2];
  const float* bv   = (const float*)d_in[3];
  const float* Wk   = (const float*)d_in[4];
  const float* bk   = (const float*)d_in[5];
  const float* Wq   = (const float*)d_in[6];
  const float* bq   = (const float*)d_in[7];
  const float* Wo   = (const float*)d_in[8];
  const float* bo   = (const float*)d_in[9];
  float* out = (float*)d_out;

  _Float16* ws = (_Float16*)d_ws;
  const size_t E = (size_t)B_ * N_ * H_;  // 8Mi elements per tensor
  _Float16* xh  = ws;
  _Float16* qh  = ws + E;
  _Float16* kh  = ws + 2 * E;
  _Float16* vth = ws + 3 * E;             // [b][h][n] transposed V
  _Float16* ah  = ws + 4 * E;
  _Float16* wqh = ws + 5 * E;
  _Float16* wkh = wqh + DIN_ * H_;
  _Float16* wvh = wkh + DIN_ * H_;
  _Float16* woh = wvh + DIN_ * H_;

  cvt_f32_to_f16<<<1024, 256, 0, stream>>>(x, xh, (int)E);
  cvt_transpose_w<<<256, 256, 0, stream>>>(Wq, wqh);
  cvt_transpose_w<<<256, 256, 0, stream>>>(Wk, wkh);
  cvt_transpose_w<<<256, 256, 0, stream>>>(Wv, wvh);
  cvt_transpose_w<<<256, 256, 0, stream>>>(Wo, woh);

  dim3 gProj(B_ * N_ / 16, 3);
  proj_qkv<<<gProj, 256, 0, stream>>>(xh, wqh, bq, wkh, bk, wvh, bv,
                                      qh, kh, vth);

  dim3 gAtt(N_ / 16, B_);
  attention<<<gAtt, 256, 0, stream>>>(qh, kh, vth, mask, ah);

  out_proj<<<B_ * N_ / 16, 256, 0, stream>>>(ah, woh, bo, out);
}
